// GNNConvDropoutGlobalAttention_16647293239656
// MI455X (gfx1250) — matrix operand order repaired and checked by reading, not compiled, not run on hardware
//
#include <hip/hip_runtime.h>
#include <hip/hip_bf16.h>

#define N_NODES 100000
#define N_EDGES 1600000
#define FIN     128
#define HID     64
#define NG      64
#define NC      5

typedef __attribute__((ext_vector_type(16))) __bf16 v16bf;
typedef __attribute__((ext_vector_type(8)))  float  v8f;
typedef __attribute__((ext_vector_type(4)))  float  f4;

__device__ __forceinline__ void atomicMaxFloat(float* addr, float v) {
  if (v >= 0.0f) atomicMax((int*)addr, __float_as_int(v));
  else           atomicMin((unsigned int*)addr, __float_as_uint(v));
}

// ---------------- init: deg=1 (self loop), gmax=-inf, denom=0, pooled=0 ----
__global__ void k_init(float* deg, float* gmax, float* denom, float* pooled) {
  int i = blockIdx.x * blockDim.x + threadIdx.x;
  if (i < N_NODES) deg[i] = 1.0f;
  if (i < NG) { gmax[i] = -__builtin_inff(); denom[i] = 0.0f; }
  if (i < NG * HID) pooled[i] = 0.0f;
}

// ---------------- degree accumulation over edge destinations ---------------
__global__ void k_deg(const int* __restrict__ dst, float* __restrict__ deg) {
  int e = blockIdx.x * blockDim.x + threadIdx.x;
  if (e < N_EDGES) unsafeAtomicAdd(&deg[dst[e]], 1.0f);
}

// ---------------- dinv = rsqrt(deg), in place -------------------------------
__global__ void k_dinv(float* deg) {
  int i = blockIdx.x * blockDim.x + threadIdx.x;
  if (i < N_NODES) deg[i] = rsqrtf(deg[i]);
}

// ---- pre-swizzle W_conv (f32, [128][64]) into per-lane WMMA B fragments ----
// fragment index = ((t*4 + c)*32 + lane)*16 + e ; K = t*32 + (lane>>4)*16 + e,
// N = c*16 + (lane&15)  (contiguous k order, same as the A fragments in k_gemm)
__global__ void k_wfrag(const float* __restrict__ W, __bf16* __restrict__ wf) {
  int tid = blockIdx.x * blockDim.x + threadIdx.x;
  if (tid >= 4 * 4 * 32 * 16) return;
  int e    = tid & 15;
  int lane = (tid >> 4) & 31;
  int c    = (tid >> 9) & 3;
  int t    = tid >> 11;
  int k = t * 32 + ((lane >> 4) * 16) + e;
  int n = c * 16 + (lane & 15);
  wf[tid] = (__bf16)W[k * HID + n];
}

// ---------------- xw = x @ W_conv via v_wmma_f32_16x16x32_bf16 --------------
// 4 waves / block; each wave computes a 16-row x 64-col tile (4 col tiles).
__global__ void __launch_bounds__(128) k_gemm(const float* __restrict__ x,
                                              const __bf16* __restrict__ wfrag,
                                              float* __restrict__ xw) {
  const int lane  = threadIdx.x & 31;
  const int wave  = threadIdx.x >> 5;
  const int r0    = (blockIdx.x * 4 + wave) * 16;
  const int m     = lane & 15;
  const int hi    = lane >> 4;     // 0 or 1
  const int abase = hi * 16;       // A-fragment K sub-offset
  const v16bf* Bf = (const v16bf*)wfrag;

  v8f acc0 = {}, acc1 = {}, acc2 = {}, acc3 = {};

  int row = r0 + m;
  if (row > N_NODES - 1) row = N_NODES - 1;      // clamp, store predicated later
  const float* p = x + (size_t)row * FIN;

  #pragma unroll
  for (int t = 0; t < 4; ++t) {
    const int koff = t * 32;
    // A 16x32 bf16 per-lane layout: elems 0..7 -> K=koff+abase+e,
    //                               elems 8..15 -> K=koff+abase+8+(e-8)
    v8f c0 = *(const v8f*)(p + koff + abase);
    v8f c1 = *(const v8f*)(p + koff + abase + 8);
    v16bf a;
    #pragma unroll
    for (int e = 0; e < 8; ++e) {
      a[e]     = (__bf16)c0[e];
      a[8 + e] = (__bf16)c1[e];
    }
    v16bf b0 = Bf[(t * 4 + 0) * 32 + lane];
    v16bf b1 = Bf[(t * 4 + 1) * 32 + lane];
    v16bf b2 = Bf[(t * 4 + 2) * 32 + lane];
    v16bf b3 = Bf[(t * 4 + 3) * 32 + lane];
    acc0 = __builtin_amdgcn_wmma_f32_16x16x32_bf16(false, a, false, b0, (short)0, acc0, false, false);
    acc1 = __builtin_amdgcn_wmma_f32_16x16x32_bf16(false, a, false, b1, (short)0, acc1, false, false);
    acc2 = __builtin_amdgcn_wmma_f32_16x16x32_bf16(false, a, false, b2, (short)0, acc2, false, false);
    acc3 = __builtin_amdgcn_wmma_f32_16x16x32_bf16(false, a, false, b3, (short)0, acc3, false, false);
  }

  // C/D layout: VGPR j -> M = j + 8*hi, N = (lane&15) + 16*c
  const int n     = lane & 15;
  const int rbase = r0 + hi * 8;
  #pragma unroll
  for (int j = 0; j < 8; ++j) {
    int r = rbase + j;
    if (r < N_NODES) {
      float* o = xw + (size_t)r * HID + n;
      o[0]  = acc0[j];
      o[16] = acc1[j];
      o[32] = acc2[j];
      o[48] = acc3[j];
    }
  }
}

// ---------------- h init with self-loop term: h = dinv^2 * xw ---------------
__global__ void k_self(const float* __restrict__ xw, const float* __restrict__ dinv,
                       float* __restrict__ h) {
  int i = blockIdx.x * blockDim.x + threadIdx.x;
  if (i >= N_NODES * HID) return;
  float d = dinv[i >> 6];
  h[i] = d * d * xw[i];
}

// ---------------- edge scatter: h[dst] += dinv[s]*dinv[d] * xw[src] ---------
// 16 threads per edge, 4 features each (xw/h are L2-resident: 25.6MB << 192MB)
__global__ void k_edges(const int* __restrict__ ei, const float* __restrict__ xw,
                        const float* __restrict__ dinv, float* __restrict__ h) {
  int tid = blockIdx.x * blockDim.x + threadIdx.x;
  if (tid >= N_EDGES * 16) return;
  int e = tid >> 4;
  int q = tid & 15;
  int s = ei[e];
  int d = ei[N_EDGES + e];
  float norm = dinv[s] * dinv[d];
  f4 xs = *(const f4*)(xw + (size_t)s * HID + q * 4);
  float* hp = h + (size_t)d * HID + q * 4;
  unsafeAtomicAdd(hp + 0, norm * xs.x);
  unsafeAtomicAdd(hp + 1, norm * xs.y);
  unsafeAtomicAdd(hp + 2, norm * xs.z);
  unsafeAtomicAdd(hp + 3, norm * xs.w);
}

// ------- bias + relu (in place) + gate = h@gate_w + gate_b, segment max -----
__global__ void k_gate(float* __restrict__ h, const float* __restrict__ b_conv,
                       const float* __restrict__ gate_w, const float* __restrict__ gate_b,
                       const int* __restrict__ batch, float* __restrict__ gate,
                       float* __restrict__ gmax) {
  int n = blockIdx.x * blockDim.x + threadIdx.x;
  if (n >= N_NODES) return;
  float* hp = h + (size_t)n * HID;
  float acc = gate_b[0];
  #pragma unroll 8
  for (int f = 0; f < HID; ++f) {
    float v = hp[f] + b_conv[f];
    v = v > 0.0f ? v : 0.0f;
    hp[f] = v;
    acc += v * gate_w[f];
  }
  gate[n] = acc;
  atomicMaxFloat(&gmax[batch[n]], acc);
}

// ---------------- e = exp(gate - gmax[b]); denom[b] += e --------------------
__global__ void k_exp(const int* __restrict__ batch, const float* __restrict__ gmax,
                      float* __restrict__ gate, float* __restrict__ denom) {
  int n = blockIdx.x * blockDim.x + threadIdx.x;
  if (n >= N_NODES) return;
  int b = batch[n];
  float ev = __expf(gate[n] - gmax[b]);
  gate[n] = ev;
  unsafeAtomicAdd(&denom[b], ev);
}

// ------- pooled[b] += (e/denom[b]) * h[n] ; LDS-staged per block ------------
__global__ void __launch_bounds__(256) k_pool(const int* __restrict__ batch,
                                              const float* __restrict__ gate,
                                              const float* __restrict__ denom,
                                              const float* __restrict__ h,
                                              float* __restrict__ pooled) {
  __shared__ float acc[NG * HID];            // 16 KB of the 320 KB WGP LDS
  for (int j = threadIdx.x; j < NG * HID; j += 256) acc[j] = 0.0f;
  __syncthreads();
  int n = blockIdx.x * 256 + threadIdx.x;
  if (n < N_NODES) {
    int b = batch[n];
    float w = gate[n] / denom[b];
    const float* hp = h + (size_t)n * HID;
    #pragma unroll 8
    for (int f = 0; f < HID; ++f)
      atomicAdd(&acc[b * HID + f], w * hp[f]);   // ds_add_f32
  }
  __syncthreads();
  for (int j = threadIdx.x; j < NG * HID; j += 256) {
    float v = acc[j];
    if (v != 0.0f) unsafeAtomicAdd(&pooled[j], v);
  }
}

// ---------------- head: x_lin1 = relu(pooled@W1+b1); logits; softmax --------
__global__ void __launch_bounds__(256) k_head(const float* __restrict__ pooled,
                                              const float* __restrict__ W1,
                                              const float* __restrict__ b1,
                                              const float* __restrict__ W2,
                                              const float* __restrict__ b2,
                                              const float* __restrict__ sex,
                                              const float* __restrict__ cag,
                                              float* __restrict__ out) {
  __shared__ float xl[NG * HID];
  __shared__ float lg[NG * NC];
  for (int idx = threadIdx.x; idx < NG * HID; idx += 256) {
    int g = idx >> 6, j = idx & 63;
    float s = b1[j];
    const float* pr = pooled + g * HID;
    #pragma unroll 8
    for (int k = 0; k < HID; ++k) s += pr[k] * W1[k * HID + j];
    s = s > 0.0f ? s : 0.0f;
    xl[idx] = s;
    out[NG * NC + idx] = s;                 // x_lin1 (second tuple output)
  }
  __syncthreads();
  for (int idx = threadIdx.x; idx < NG * NC; idx += 256) {
    int g = idx / NC, c = idx % NC;
    float s = b2[c];
    #pragma unroll 8
    for (int j = 0; j < HID; ++j) s += xl[g * HID + j] * W2[j * NC + c];
    s += sex[g] * W2[HID * NC + c] + cag[g] * W2[(HID + 1) * NC + c];
    lg[idx] = s;
  }
  __syncthreads();
  if (threadIdx.x < NG) {
    int g = threadIdx.x;
    float mx = lg[g * NC];
    for (int c = 1; c < NC; ++c) mx = fmaxf(mx, lg[g * NC + c]);
    float ex[NC], sum = 0.0f;
    for (int c = 0; c < NC; ++c) { ex[c] = __expf(lg[g * NC + c] - mx); sum += ex[c]; }
    for (int c = 0; c < NC; ++c) out[g * NC + c] = ex[c] / sum;   // prediction
  }
}

extern "C" void kernel_launch(void* const* d_in, const int* in_sizes, int n_in,
                              void* d_out, int out_size, void* d_ws, size_t ws_size,
                              hipStream_t stream) {
  const float* x      = (const float*)d_in[0];
  const int*   ei     = (const int*)d_in[1];
  const int*   batch  = (const int*)d_in[2];
  const float* sex    = (const float*)d_in[3];
  const float* cag    = (const float*)d_in[4];
  const float* W_conv = (const float*)d_in[5];
  const float* b_conv = (const float*)d_in[6];
  const float* gate_w = (const float*)d_in[7];
  const float* gate_b = (const float*)d_in[8];
  const float* W1     = (const float*)d_in[9];
  const float* b1     = (const float*)d_in[10];
  const float* W2     = (const float*)d_in[11];
  const float* b2     = (const float*)d_in[12];
  float* out = (float*)d_out;
  (void)in_sizes; (void)n_in; (void)out_size; (void)ws_size;

  char* ws = (char*)d_ws;
  size_t off = 0;
  auto salloc = [&](size_t bytes) -> void* {
    void* p = ws + off;
    off += (bytes + 255) & ~(size_t)255;
    return p;
  };
  float*  deg    = (float*)salloc((size_t)N_NODES * 4);            // reused as dinv
  float*  xw     = (float*)salloc((size_t)N_NODES * HID * 4);
  float*  h      = (float*)salloc((size_t)N_NODES * HID * 4);
  float*  gate   = (float*)salloc((size_t)N_NODES * 4);            // reused as e
  float*  gmax   = (float*)salloc((size_t)NG * 4);
  float*  denom  = (float*)salloc((size_t)NG * 4);
  float*  pooled = (float*)salloc((size_t)NG * HID * 4);
  __bf16* wfrag  = (__bf16*)salloc((size_t)4 * 4 * 32 * 16 * 2);

  k_init <<<(N_NODES + 255) / 256, 256, 0, stream>>>(deg, gmax, denom, pooled);
  k_deg  <<<(N_EDGES + 255) / 256, 256, 0, stream>>>(ei + N_EDGES, deg);
  k_dinv <<<(N_NODES + 255) / 256, 256, 0, stream>>>(deg);
  k_wfrag<<<(4 * 4 * 32 * 16) / 256, 256, 0, stream>>>(W_conv, wfrag);
  k_gemm <<<(N_NODES + 63) / 64, 128, 0, stream>>>(x, wfrag, xw);
  k_self <<<(N_NODES * HID + 255) / 256, 256, 0, stream>>>(xw, deg, h);
  k_edges<<<(N_EDGES * 16 + 255) / 256, 256, 0, stream>>>(ei, xw, deg, h);
  k_gate <<<(N_NODES + 255) / 256, 256, 0, stream>>>(h, b_conv, gate_w, gate_b, batch, gate, gmax);
  k_exp  <<<(N_NODES + 255) / 256, 256, 0, stream>>>(batch, gmax, gate, denom);
  k_pool <<<(N_NODES + 255) / 256, 256, 0, stream>>>(batch, gate, denom, h, pooled);
  k_head <<<1, 256, 0, stream>>>(pooled, W1, b1, W2, b2, sex, cag, out);
}
